// Cell2Cell_4750233829569
// MI455X (gfx1250) — matrix operand order repaired and checked by reading, not compiled, not run on hardware
//
#include <hip/hip_runtime.h>
#include <cstdint>
#include <cstddef>

#define N_PTS 4096
#define IN_SZ 2048
#define HID   256
#define HEADS 4
#define K_KNN 30
#define KA    10

typedef __attribute__((ext_vector_type(16))) __bf16 v16bf;
typedef __attribute__((ext_vector_type(8)))  __bf16 v8bf;
typedef __attribute__((ext_vector_type(8)))  float  v8f;

// ---------------------------------------------------------------------------
// Fragment loaders (wave32 WMMA layouts, CDNA5 ISA §7.12.2)
// A: 16x32 bf16, lane (m=lane&15, half=lane>>4): VGPR0-3 = K[half*8 .. +8),
//    VGPR4-7 = K[16+half*8 .. +8)  -> two contiguous b128 loads per lane.
// B: from BT (row-major [Ncols x K]); lane (n=lane&15, half=lane>>4)
//    holds column n, K[half*8 .. +8) and K[16+half*8 .. +8) -> two b128 loads per lane.
// ---------------------------------------------------------------------------
__device__ __forceinline__ v16bf load_a_frag(const __bf16* __restrict__ A,
                                             int lda, int row0, int kb) {
  int lane = threadIdx.x & 31;
  int m = lane & 15, half = lane >> 4;
  const __bf16* p = A + (size_t)(row0 + m) * lda + kb + half * 8;
  v8bf lo = *(const v8bf*)p;
  v8bf hi = *(const v8bf*)(p + 16);
  v16bf a;
#pragma unroll
  for (int t = 0; t < 8; ++t) { a[t] = lo[t]; a[t + 8] = hi[t]; }
  return a;
}

__device__ __forceinline__ v16bf load_b_frag(const __bf16* __restrict__ BT,
                                             int ldb, int col0, int kb) {
  int lane = threadIdx.x & 31;
  int n = lane & 15, half = lane >> 4;
  const __bf16* p = BT + (size_t)(col0 + n) * ldb + kb + half * 8;
  v8bf lo = *(const v8bf*)p;
  v8bf hi = *(const v8bf*)(p + 16);
  v16bf b;
#pragma unroll
  for (int t = 0; t < 8; ++t) { b[t] = lo[t]; b[t + 8] = hi[t]; }
  return b;
}

// ---------------------------------------------------------------------------
// Generic bf16 GEMM: C[M x Ncols] = A[M x K] * BT[Ncols x K]^T
// 128 threads (4 waves), each wave a 64x64 tile via 4x4 v_wmma_f32_16x16x32_bf16.
// MODE is a compile-time template parameter so each instantiation gets a
// minimal epilogue (avoids the accumulator->scratch spill seen with the
// runtime-branching version):
//   0: dist    : Cf = sqrt(max(rown[r]+coln[c]-2*acc,0)+1e-10)
//   1: bf16    : Cb = bf16(acc + bias[c])
//   2: bf16    : Cb = bf16(acc)
//   3: bf16+T  : Cb = bf16(acc), CTb[c][r] = bf16(acc)
//   4: accum   : Cf += scale * acc
// __launch_bounds__(128, 1): allow the full wave32 VGPR budget (no spills).
// All launch dims are exact multiples of the tile sizes (no bounds checks).
// ---------------------------------------------------------------------------
template <int MODE>
__global__ __launch_bounds__(128, 1) void wmma_gemm_bf16(
    const __bf16* __restrict__ A, const __bf16* __restrict__ BT,
    int lda, int ldb, int K,
    float* __restrict__ Cf, int ldc,
    const float* __restrict__ rown, const float* __restrict__ coln,
    __bf16* __restrict__ Cb, __bf16* __restrict__ CTb, int ldcb, int ldct,
    const float* __restrict__ bias, float scale) {
  int wid  = threadIdx.x >> 5;
  int lane = threadIdx.x & 31;
  int row0 = blockIdx.y * 128 + (wid >> 1) * 64;
  int col0 = blockIdx.x * 128 + (wid & 1) * 64;

  v8f acc[4][4] = {};

  for (int kb = 0; kb < K; kb += 32) {
    v16bf af[4], bfr[4];
#pragma unroll
    for (int i = 0; i < 4; ++i) af[i] = load_a_frag(A, lda, row0 + i * 16, kb);
#pragma unroll
    for (int j = 0; j < 4; ++j) bfr[j] = load_b_frag(BT, ldb, col0 + j * 16, kb);
#pragma unroll
    for (int i = 0; i < 4; ++i)
#pragma unroll
      for (int j = 0; j < 4; ++j)
        acc[i][j] = __builtin_amdgcn_wmma_f32_16x16x32_bf16(
            false, af[i], false, bfr[j], (short)0, acc[i][j], false, false);
  }

  // C/D layout: lane (n=lane&15), rows mo..mo+7 with mo=(lane>>4)*8 in VGPRs 0..7
  int n  = lane & 15;
  int mo = (lane >> 4) * 8;
#pragma unroll
  for (int i = 0; i < 4; ++i) {
#pragma unroll
    for (int j = 0; j < 4; ++j) {
#pragma unroll
      for (int r = 0; r < 8; ++r) {
        int row = row0 + i * 16 + mo + r;
        int col = col0 + j * 16 + n;
        float v = acc[i][j][r];
        if (MODE == 0) {
          float sq = rown[row] + coln[col] - 2.0f * v;
          Cf[(size_t)row * ldc + col] = sqrtf(fmaxf(sq, 0.0f) + 1e-10f);
        } else if (MODE == 1) {
          Cb[(size_t)row * ldcb + col] = (__bf16)(v + bias[col]);
        } else if (MODE == 2) {
          Cb[(size_t)row * ldcb + col] = (__bf16)v;
        } else if (MODE == 3) {
          __bf16 bv = (__bf16)v;
          Cb[(size_t)row * ldcb + col] = bv;
          CTb[(size_t)col * ldct + row] = bv;
        } else {
          Cf[(size_t)row * ldc + col] += scale * v;
        }
      }
    }
  }
}

// ---------------------------------------------------------------------------
// Block reduction helper (blockDim.x == 256)
// ---------------------------------------------------------------------------
__device__ __forceinline__ float blk_sum(float v, float* red) {
  int tid = threadIdx.x;
  red[tid] = v; __syncthreads();
  for (int s = 128; s > 0; s >>= 1) {
    if (tid < s) red[tid] += red[tid + s];
    __syncthreads();
  }
  float r = red[0]; __syncthreads();
  return r;
}

// ---------------------------------------------------------------------------
// Per-row: adaptive bandwidth (11th smallest dist), aff=exp(-(d/md)^2),
// prune to top-30 (>= 30th largest), diag=1. In-place on dist buffer.
// kth order statistics via 32-step binary search on float bit patterns
// (all values > 0, so uint compare is monotone). Row lives in LDS (16 KB).
// ---------------------------------------------------------------------------
__global__ __launch_bounds__(256) void knn_select_kernel(float* __restrict__ dist) {
  __shared__ float srow[N_PTS];
  __shared__ float red[256];
  int row = blockIdx.x, tid = threadIdx.x;
  float* drow = dist + (size_t)row * N_PTS;
  for (int j = tid; j < N_PTS; j += 256) srow[j] = drow[j];
  __syncthreads();

  // 0-based order statistic KA (11th smallest)
  unsigned lo = 0u, hi = 0xffffffffu;
  for (int it = 0; it < 32; ++it) {
    unsigned mid = lo + ((hi - lo) >> 1);
    int c = 0;
    for (int j = tid; j < N_PTS; j += 256)
      c += (__float_as_uint(srow[j]) <= mid) ? 1 : 0;
    float tot = blk_sum((float)c, red);
    if (tot >= (float)(KA + 1)) hi = mid; else lo = mid + 1;
  }
  float md = __uint_as_float(lo) + 1e-10f;

  for (int j = tid; j < N_PTS; j += 256) {
    float dv = srow[j] / md;
    srow[j] = __expf(-dv * dv);
  }
  __syncthreads();

  // 0-based order statistic (N - K_KNN) of aff = 30th largest
  lo = 0u; hi = 0xffffffffu;
  for (int it = 0; it < 32; ++it) {
    unsigned mid = lo + ((hi - lo) >> 1);
    int c = 0;
    for (int j = tid; j < N_PTS; j += 256)
      c += (__float_as_uint(srow[j]) <= mid) ? 1 : 0;
    float tot = blk_sum((float)c, red);
    if (tot >= (float)(N_PTS - K_KNN + 1)) hi = mid; else lo = mid + 1;
  }
  float kth = __uint_as_float(lo);

  for (int j = tid; j < N_PTS; j += 256) {
    float a = srow[j];
    a = (a >= kth) ? a : 0.0f;
    if (j == row) a = 1.0f;
    drow[j] = a;
  }
}

// ---------------------------------------------------------------------------
// Per-row: S = aff + aff^T, softmax over the row, emit bf16 P and P^T.
// ---------------------------------------------------------------------------
__global__ __launch_bounds__(256) void sym_softmax_kernel(
    const float* __restrict__ aff, __bf16* __restrict__ P,
    __bf16* __restrict__ PT) {
  __shared__ float s[N_PTS];
  __shared__ float red[256];
  int row = blockIdx.x, tid = threadIdx.x;
  const float* arow = aff + (size_t)row * N_PTS;
  for (int j = tid; j < N_PTS; j += 256)
    s[j] = arow[j] + aff[(size_t)j * N_PTS + row];
  __syncthreads();

  float m = -3.4e38f;
  for (int j = tid; j < N_PTS; j += 256) m = fmaxf(m, s[j]);
  red[tid] = m; __syncthreads();
  for (int st = 128; st > 0; st >>= 1) {
    if (tid < st) red[tid] = fmaxf(red[tid], red[tid + st]);
    __syncthreads();
  }
  float rmax = red[0]; __syncthreads();

  float sum = 0.0f;
  for (int j = tid; j < N_PTS; j += 256) {
    float e = __expf(s[j] - rmax);
    s[j] = e;
    sum += e;
  }
  float tot = blk_sum(sum, red);
  float inv = 1.0f / tot;

  for (int j = tid; j < N_PTS; j += 256) {
    __bf16 pv = (__bf16)(s[j] * inv);
    P[(size_t)row * N_PTS + j]  = pv;
    PT[(size_t)j * N_PTS + row] = pv;
  }
}

// ---------------------------------------------------------------------------
// Row squared-norms of a [N_PTS x HID] bf16 matrix (HID == blockDim.x == 256)
// ---------------------------------------------------------------------------
__global__ __launch_bounds__(256) void rownorm_kernel(
    const __bf16* __restrict__ q, float* __restrict__ outn) {
  __shared__ float red[256];
  int row = blockIdx.x, tid = threadIdx.x;
  float v = (float)q[(size_t)row * HID + tid];
  red[tid] = v * v; __syncthreads();
  for (int st = 128; st > 0; st >>= 1) {
    if (tid < st) red[tid] += red[tid + st];
    __syncthreads();
  }
  if (tid == 0) outn[row] = red[0];
}

// ---------------------------------------------------------------------------
// Conversions / init
// ---------------------------------------------------------------------------
__global__ void cvt_x_kernel(const float* __restrict__ X,
                             __bf16* __restrict__ Xb,
                             __bf16* __restrict__ XTb) {
  size_t idx = (size_t)blockIdx.x * blockDim.x + threadIdx.x;
  if (idx >= (size_t)N_PTS * IN_SZ) return;
  int i = (int)(idx >> 11);       // / IN_SZ (2048)
  int j = (int)(idx & 2047);
  __bf16 v = (__bf16)X[idx];
  Xb[idx] = v;
  XTb[(size_t)j * N_PTS + i] = v;
}

__global__ void cvt_kernel(const float* __restrict__ in,
                           __bf16* __restrict__ outp, size_t n) {
  size_t idx = (size_t)blockIdx.x * blockDim.x + threadIdx.x;
  if (idx < n) outp[idx] = (__bf16)in[idx];
}

__global__ void zero_kernel(float* __restrict__ p, size_t n) {
  size_t idx = (size_t)blockIdx.x * blockDim.x + threadIdx.x;
  if (idx < n) p[idx] = 0.0f;
}

// ---------------------------------------------------------------------------
// Launcher
// ---------------------------------------------------------------------------
extern "C" void kernel_launch(void* const* d_in, const int* in_sizes, int n_in,
                              void* d_out, int out_size, void* d_ws,
                              size_t ws_size, hipStream_t stream) {
  (void)in_sizes; (void)n_in; (void)out_size; (void)ws_size;
  const float* X  = (const float*)d_in[0];
  const float* Wq = (const float*)d_in[1];
  const float* bq = (const float*)d_in[2];
  const float* Wk = (const float*)d_in[3];
  const float* bk = (const float*)d_in[4];
  float* out = (float*)d_out;

  char* w = (char*)d_ws;
  size_t off = 0;
  auto alloc = [&](size_t bytes) -> void* {
    off = (off + 255) & ~(size_t)255;
    void* p = w + off;
    off += bytes;
    return p;
  };

  const size_t ND  = (size_t)N_PTS * IN_SZ;
  const size_t NN  = (size_t)N_PTS * N_PTS;
  const size_t NH  = (size_t)N_PTS * HID;
  const size_t WSZ = (size_t)HEADS * HID * IN_SZ;

  __bf16* Xb   = (__bf16*)alloc(ND * 2);
  __bf16* XTb  = (__bf16*)alloc(ND * 2);
  __bf16* Wqb  = (__bf16*)alloc(WSZ * 2);
  __bf16* Wkb  = (__bf16*)alloc(WSZ * 2);
  __bf16* qb   = (__bf16*)alloc(NH * 2);
  __bf16* kbuf = (__bf16*)alloc(NH * 2);
  float*  qq   = (float*)alloc((size_t)N_PTS * 4);
  float*  kk   = (float*)alloc((size_t)N_PTS * 4);
  float*  aff  = (float*)alloc(NN * 4);
  __bf16* Pb   = (__bf16*)alloc(NN * 2);
  __bf16* PTb  = (__bf16*)alloc(NN * 2);
  __bf16* P2b  = (__bf16*)alloc(NN * 2);
  __bf16* P2Tb = (__bf16*)alloc(NN * 2);
  __bf16* P4b  = (__bf16*)alloc(NN * 2);
  __bf16* P6b  = (__bf16*)alloc(NN * 2);

  zero_kernel<<<(unsigned)((ND + 255) / 256), 256, 0, stream>>>(out, ND);
  cvt_x_kernel<<<(unsigned)((ND + 255) / 256), 256, 0, stream>>>(X, Xb, XTb);
  cvt_kernel<<<(unsigned)((WSZ + 255) / 256), 256, 0, stream>>>(Wq, Wqb, WSZ);
  cvt_kernel<<<(unsigned)((WSZ + 255) / 256), 256, 0, stream>>>(Wk, Wkb, WSZ);

  dim3 blk(128);
  dim3 gq(HID / 128, N_PTS / 128);      // 2 x 32
  dim3 gnn(N_PTS / 128, N_PTS / 128);   // 32 x 32
  dim3 gctx(IN_SZ / 128, N_PTS / 128);  // 16 x 32

  for (int h = 0; h < HEADS; ++h) {
    const __bf16* Wqh = Wqb + (size_t)h * HID * IN_SZ;
    const __bf16* Wkh = Wkb + (size_t)h * HID * IN_SZ;
    // q = X * Wq_h^T + bq_h   (bf16 out)
    wmma_gemm_bf16<1><<<gq, blk, 0, stream>>>(Xb, Wqh, IN_SZ, IN_SZ, IN_SZ,
        nullptr, 0, nullptr, nullptr, qb, nullptr, HID, 0, bq + h * HID, 0.f);
    // k = X * Wk_h^T + bk_h
    wmma_gemm_bf16<1><<<gq, blk, 0, stream>>>(Xb, Wkh, IN_SZ, IN_SZ, IN_SZ,
        nullptr, 0, nullptr, nullptr, kbuf, nullptr, HID, 0, bk + h * HID, 0.f);

    rownorm_kernel<<<N_PTS, 256, 0, stream>>>(qb, qq);
    rownorm_kernel<<<N_PTS, 256, 0, stream>>>(kbuf, kk);

    // dist = sqrt(max(|q|^2 + |k|^2 - 2 q.k, 0) + 1e-10)  (f32 into aff)
    wmma_gemm_bf16<0><<<gnn, blk, 0, stream>>>(qb, kbuf, HID, HID, HID,
        aff, N_PTS, qq, kk, nullptr, nullptr, 0, 0, nullptr, 0.f);

    knn_select_kernel<<<N_PTS, 256, 0, stream>>>(aff);
    sym_softmax_kernel<<<N_PTS, 256, 0, stream>>>(aff, Pb, PTb);

    // P2 = P * P  (emit P2 and P2^T)
    wmma_gemm_bf16<3><<<gnn, blk, 0, stream>>>(Pb, PTb, N_PTS, N_PTS, N_PTS,
        nullptr, 0, nullptr, nullptr, P2b, P2Tb, N_PTS, N_PTS, nullptr, 0.f);
    // P4 = P2 * P2
    wmma_gemm_bf16<2><<<gnn, blk, 0, stream>>>(P2b, P2Tb, N_PTS, N_PTS, N_PTS,
        nullptr, 0, nullptr, nullptr, P4b, nullptr, N_PTS, 0, nullptr, 0.f);
    // P6 = P4 * P2
    wmma_gemm_bf16<2><<<gnn, blk, 0, stream>>>(P4b, P2Tb, N_PTS, N_PTS, N_PTS,
        nullptr, 0, nullptr, nullptr, P6b, nullptr, N_PTS, 0, nullptr, 0.f);
    // out += 0.25 * P6 * X
    wmma_gemm_bf16<4><<<gctx, blk, 0, stream>>>(P6b, XTb, N_PTS, N_PTS, N_PTS,
        out, IN_SZ, nullptr, nullptr, nullptr, nullptr, 0, 0, nullptr, 0.25f);
  }
}
